// MultiTimescaleEmbedding_77833397338705
// MI455X (gfx1250) — compile-verified, matrix operand order checked by reading, not run on hardware
//
#include <hip/hip_runtime.h>
#include <hip/hip_bf16.h>

typedef __bf16 v16bf __attribute__((ext_vector_type(16)));
typedef __bf16 v8bf  __attribute__((ext_vector_type(8)));
typedef float  v8f   __attribute__((ext_vector_type(8)));

union BFPack { v16bf v; v8bf h[2]; };

#define EMB 256
#define CAP 64
#define NCAPS 8
#define ROWS_PER_PHASE 32
#define M_PHASES 4               // 4 phases x 32 rows = 128 rows per block
#define ROWS_PER_BLOCK (ROWS_PER_PHASE * M_PHASES)

// workspace layout (bytes)
static const size_t WS_WBF  = 0;                      // 512 x 256 bf16  (W_transform)
static const size_t WS_WOUT = (size_t)512 * 256 * 2;  // 256 x 64  bf16  (W_out)
static const size_t WS_BIAS = WS_WOUT + (size_t)256 * 64 * 2; // 512 f32 (b_transform + b_temporal)

// ---------------- weight prep: f32 -> bf16, fuse biases ----------------
__global__ __launch_bounds__(256) void prep_kernel(const float* __restrict__ Wt,
                                                   const float* __restrict__ bt,
                                                   const float* __restrict__ btmp,
                                                   const float* __restrict__ Wout,
                                                   __bf16* __restrict__ wbf,
                                                   __bf16* __restrict__ woutbf,
                                                   float* __restrict__ biasc) {
  int i = blockIdx.x * 256 + threadIdx.x;
  if (i < 512 * 256) wbf[i] = (__bf16)Wt[i];          // [c*64+d][e], same flat layout as W_transform
  if (i < 256 * 64)  woutbf[i] = (__bf16)Wout[i];     // [e][d], same flat layout as W_out
  if (i < 512)       biasc[i] = bt[i] + btmp[i];      // temporal state is zero -> only biases remain
}

// 16-bit WMMA A/B fragment: two contiguous 16-byte runs (elements +0..7 and +16..23)
__device__ inline v16bf load_bf_frag(const __bf16* p) {
  BFPack u;
  u.h[0] = *(const v8bf*)(p);
  u.h[1] = *(const v8bf*)(p + 16);
  return u.v;
}
// Build fragment from f32 (LDS) memory
__device__ inline v16bf cvt_bf_frag(const float* p) {
  BFPack u;
#pragma unroll
  for (int i = 0; i < 8; ++i) u.h[0][i] = (__bf16)p[i];
#pragma unroll
  for (int i = 0; i < 8; ++i) u.h[1][i] = (__bf16)p[16 + i];
  return u.v;
}

__global__ __launch_bounds__(256) void caps_kernel(const float* __restrict__ x,
                                                   const __bf16* __restrict__ wbf,
                                                   const __bf16* __restrict__ woutbf,
                                                   const float* __restrict__ biasc,
                                                   const float* __restrict__ b_out,
                                                   float* __restrict__ out) {
  __shared__ __bf16 xs[ROWS_PER_PHASE][EMB];          // 16 KB  staged x tile (bf16)
  __shared__ float  caps[ROWS_PER_PHASE][NCAPS][CAP]; // 64 KB
  __shared__ float  sv[ROWS_PER_PHASE][CAP];          // 8 KB
  __shared__ float  vv[ROWS_PER_PHASE][CAP];          // 8 KB
  __shared__ float  blg[ROWS_PER_PHASE][NCAPS];
  __shared__ float  cw[ROWS_PER_PHASE][NCAPS];
  __shared__ float  rowscale[ROWS_PER_PHASE];

  const int tid  = threadIdx.x;
  const int wave = tid >> 5;     // 0..7  == capsule index for GEMM1
  const int lane = tid & 31;
  const int l16  = lane & 15;
  const int half = lane >> 4;    // selects K 0-7 / 8-15 halves and C rows 0-7 / 8-15
  const int k0   = half * 8;

  // staging map: each thread converts one 32-float segment per row chunk
  const int st_r  = tid >> 3;          // 0..31
  const int st_e0 = (tid & 7) * 32;    // 0,32,...,224

#pragma unroll 1
  for (int ph = 0; ph < M_PHASES; ++ph) {
    const int row0 = blockIdx.x * ROWS_PER_BLOCK + ph * ROWS_PER_PHASE;

    // ---------------- stage x tile (f32 -> bf16) into LDS ----------------
    {
      const float* src = x + (size_t)(row0 + st_r) * EMB + st_e0;
#pragma unroll
      for (int g = 0; g < 4; ++g) {
        v8bf pk;
#pragma unroll
        for (int i = 0; i < 8; ++i) pk[i] = (__bf16)src[g * 8 + i];
        *(v8bf*)&xs[st_r][st_e0 + g * 8] = pk;
      }
      if (ph + 1 < M_PHASES)   // prefetch next tile into cache (global_prefetch_b8)
        __builtin_prefetch(src + (size_t)ROWS_PER_PHASE * EMB, 0, 3);
    }
    __syncthreads();

    // ---------------- GEMM1: caps_pre[32 rows][capsule=wave][64] ----------------
    // Two 16-row sub-tiles share each B fragment (2x weight reuse).
    v8f acc[2][4] = {};
#pragma unroll
    for (int kb = 0; kb < EMB; kb += 32) {
      v16bf a0 = load_bf_frag(&xs[l16][kb + k0]);        // ds_load_b128 x2
      v16bf a1 = load_bf_frag(&xs[16 + l16][kb + k0]);
#pragma unroll
      for (int t = 0; t < 4; ++t) {
        const __bf16* pb = wbf + (size_t)(wave * CAP + t * 16 + l16) * EMB + kb + k0;
        v16bf bfrag = load_bf_frag(pb);
        acc[0][t] = __builtin_amdgcn_wmma_f32_16x16x32_bf16(
            false, a0, false, bfrag, (short)0, acc[0][t], false, false);
        acc[1][t] = __builtin_amdgcn_wmma_f32_16x16x32_bf16(
            false, a1, false, bfrag, (short)0, acc[1][t], false, false);
      }
    }
    // bias + tanh -> LDS caps[row][capsule][d]   (C layout: col=l16, row=j+8*half)
#pragma unroll
    for (int m = 0; m < 2; ++m) {
#pragma unroll
      for (int t = 0; t < 4; ++t) {
        const int d = t * 16 + l16;
        const float bias = biasc[wave * CAP + d];
#pragma unroll
        for (int j = 0; j < 8; ++j)
          caps[m * 16 + j + 8 * half][wave][d] = tanhf(acc[m][t][j] + bias);
      }
    }
    blg[tid >> 3][tid & 7] = 0.0f;   // routing logits start at 0 (32x8 == 256 threads)
    __syncthreads();

    // ---------------- routing by agreement (3 iterations, in LDS) ----------------
#pragma unroll 1
    for (int it = 0; it < 3; ++it) {
      {                                            // softmax over capsules per row (32x8 threads)
        const int r = tid >> 3, c = tid & 7;
        float mx = blg[r][0];
#pragma unroll
        for (int cc = 1; cc < 8; ++cc) mx = fmaxf(mx, blg[r][cc]);
        float ssum = 0.f;
#pragma unroll
        for (int cc = 0; cc < 8; ++cc) ssum += __expf(blg[r][cc] - mx);
        cw[r][c] = __expf(blg[r][c] - mx) / ssum;
      }
      __syncthreads();
      {                                            // s[r][d] = sum_c cw*caps  (8 d's per thread)
        const int r = tid >> 3, db = (tid & 7) * 8;
#pragma unroll
        for (int dd = 0; dd < 8; ++dd) {
          float s = 0.f;
#pragma unroll
          for (int c = 0; c < 8; ++c) s += cw[r][c] * caps[r][c][db + dd];
          sv[r][db + dd] = s;
        }
      }
      __syncthreads();
      if (tid < ROWS_PER_PHASE) {                  // squash scale per row
        float ss = 0.f;
        for (int d = 0; d < CAP; ++d) { float q = sv[tid][d]; ss += q * q; }
        float n = sqrtf(ss);
        rowscale[tid] = ss / (1.0f + ss) / (n + 1e-8f);
      }
      __syncthreads();
      {                                            // v = scale * s
        const int r = tid >> 3, db = (tid & 7) * 8;
        const float sc = rowscale[r];
#pragma unroll
        for (int dd = 0; dd < 8; ++dd) vv[r][db + dd] = sc * sv[r][db + dd];
      }
      __syncthreads();
      if (it < 2) {
        {                                          // agreement -> update logits (32x8 threads)
          const int r = tid >> 3, c = tid & 7;
          float ag = 0.f;
#pragma unroll
          for (int d = 0; d < CAP; ++d) ag += caps[r][c][d] * vv[r][d];
          blg[r][c] += ag;
        }
        __syncthreads();
      }
    }

    // ---------------- output projection: out[32][256] = v @ W_out^T + b_out ----------------
    v8f oacc[2][2] = {};
#pragma unroll
    for (int kb = 0; kb < CAP; kb += 32) {
      v16bf a0 = cvt_bf_frag(&vv[l16][kb + k0]);
      v16bf a1 = cvt_bf_frag(&vv[16 + l16][kb + k0]);
#pragma unroll
      for (int t = 0; t < 2; ++t) {
        const __bf16* pb = woutbf + (size_t)(wave * 32 + t * 16 + l16) * CAP + kb + k0;
        v16bf bfrag = load_bf_frag(pb);
        oacc[0][t] = __builtin_amdgcn_wmma_f32_16x16x32_bf16(
            false, a0, false, bfrag, (short)0, oacc[0][t], false, false);
        oacc[1][t] = __builtin_amdgcn_wmma_f32_16x16x32_bf16(
            false, a1, false, bfrag, (short)0, oacc[1][t], false, false);
      }
    }
#pragma unroll
    for (int m = 0; m < 2; ++m) {
#pragma unroll
      for (int t = 0; t < 2; ++t) {
        const int e = wave * 32 + t * 16 + l16;
        const float bo = b_out[e];
#pragma unroll
        for (int j = 0; j < 8; ++j)
          out[(size_t)(row0 + m * 16 + j + 8 * half) * EMB + e] = oacc[m][t][j] + bo;
      }
    }
    __syncthreads();  // protect LDS reuse in next phase
  }
}

extern "C" void kernel_launch(void* const* d_in, const int* in_sizes, int n_in,
                              void* d_out, int out_size, void* d_ws, size_t ws_size,
                              hipStream_t stream) {
  const float* x    = (const float*)d_in[0];
  const float* Wt   = (const float*)d_in[1];
  const float* bt   = (const float*)d_in[2];
  // d_in[3] = W_temporal: multiplied by a zero state in the reference -> unused
  const float* btmp = (const float*)d_in[4];
  const float* Wout = (const float*)d_in[5];
  const float* bo   = (const float*)d_in[6];
  float* out = (float*)d_out;

  char* ws = (char*)d_ws;
  __bf16* wbf    = (__bf16*)(ws + WS_WBF);
  __bf16* woutbf = (__bf16*)(ws + WS_WOUT);
  float*  biasc  = (float*)(ws + WS_BIAS);

  prep_kernel<<<(512 * 256 + 255) / 256, 256, 0, stream>>>(Wt, bt, btmp, Wout, wbf, woutbf, biasc);

  const int rows = in_sizes[0] / EMB;               // B = 262144
  dim3 grid(rows / ROWS_PER_BLOCK);                 // 2048 blocks
  caps_kernel<<<grid, 256, 0, stream>>>(x, wbf, woutbf, biasc, bo, out);
}
